// InfiniteHoloGPT_57569741635832
// MI455X (gfx1250) — matrix operand order repaired and checked by reading, not compiled, not run on hardware
//
#include <hip/hip_runtime.h>
#include <hip/hip_bf16.h>

// ---------------------------------------------------------------------------
// InfiniteHoloGPT forward for MI455X (gfx1250, wave32, WMMA).
// Weights + GEMM A-operands are pre-converted to bf16 so the WMMA GEMM inner
// loop is pure b128-load -> v_wmma_f32_16x16x32_bf16 (no VALU converts in the
// hot loop, loads stay in flight). fp32 accumulation; scan/LN/gates fp32.
// ---------------------------------------------------------------------------

#define BATCH   2
#define SEQ     1024
#define NTOK    (BATCH * SEQ)      // 2048
#define EMB     512
#define HEADS   16
#define HDIM    32
#define FFN     2048
#define VOCAB   32000

typedef __bf16 bf16;
typedef __attribute__((ext_vector_type(16))) __bf16 v16bf;
typedef __attribute__((ext_vector_type(8)))  float  v8f;

// ------------------------------- helpers -----------------------------------

__device__ __forceinline__ float wave_sum(float v) {
#pragma unroll
  for (int off = 16; off > 0; off >>= 1) v += __shfl_xor(v, off, 32);
  return v;
}

// block of 256 threads (8 waves)
__device__ __forceinline__ float block_sum256(float v, float* sm) {
  v = wave_sum(v);
  const int lane = threadIdx.x & 31, w = threadIdx.x >> 5;
  if (lane == 0) sm[w] = v;
  __syncthreads();
  float t = (threadIdx.x < 8) ? sm[threadIdx.x] : 0.f;
  if (w == 0) {
#pragma unroll
    for (int off = 4; off > 0; off >>= 1) t += __shfl_xor(t, off, 32);
    if (lane == 0) sm[8] = t;
  }
  __syncthreads();
  float r = sm[8];
  __syncthreads();
  return r;
}

__device__ __forceinline__ float sigmoidf(float x) { return 1.f / (1.f + expf(-x)); }
__device__ __forceinline__ float gelu_exact(float x) {
  return 0.5f * x * (1.f + erff(x * 0.7071067811865475f));
}

// 16 contiguous bf16 -> fragment (two b128 loads); p must be 32B aligned
__device__ __forceinline__ v16bf load_frag16(const bf16* p) {
  union { float4 f[2]; v16bf v; } u;
  const float4* q = (const float4*)p;
  u.f[0] = q[0];
  u.f[1] = q[1];
  return u.v;
}
// A fragment: 8 bf16 at p, 8 bf16 at p+16 elements (two b128 loads)
__device__ __forceinline__ v16bf load_fragA(const bf16* p) {
  union { float4 f[2]; v16bf v; } u;
  u.f[0] = *(const float4*)p;
  u.f[1] = *(const float4*)(p + 16);
  return u.v;
}

// ------------------------- fp32 -> bf16 conversion --------------------------
__global__ __launch_bounds__(256)
void cvt_bf16_kernel(const float* __restrict__ in, bf16* __restrict__ out, int n4) {
  int i = blockIdx.x * 256 + threadIdx.x;   // over n/4
  if (i < n4) {
    float4 f = ((const float4*)in)[i];
    union { bf16 h[4]; float2 f2; } u;
    u.h[0] = (bf16)f.x; u.h[1] = (bf16)f.y; u.h[2] = (bf16)f.z; u.h[3] = (bf16)f.w;
    ((float2*)out)[i] = u.f2;
  }
}

// ------------------------------ embedding ----------------------------------
// x = ((c - softplus(o)) + (c + softplus(o))) / 2
__global__ __launch_bounds__(256)
void embed_kernel(const int* __restrict__ idx, const float* __restrict__ center,
                  const float* __restrict__ offset, float* __restrict__ x) {
  int i = blockIdx.x * 256 + threadIdx.x;          // over NTOK*EMB
  int tok = i >> 9, e = i & (EMB - 1);
  int id = idx[tok];
  size_t p = (size_t)id * EMB + e;
  float c = center[p];
  float o = offset[p];
  float sp = log1pf(expf(o));
  x[i] = 0.5f * ((c - sp) + (c + sp));
}

// ------------------------------ layernorm ----------------------------------
// one block (256 thr) per token; E = 512 -> 2 elements / thread; bf16 output
__global__ __launch_bounds__(256)
void ln_kernel(const float* __restrict__ in, bf16* __restrict__ out,
               const float* __restrict__ g, const float* __restrict__ b) {
  __shared__ float sm[9];
  const int t = blockIdx.x;
  const float* xr = in + (size_t)t * EMB;
  float v0 = xr[threadIdx.x], v1 = xr[threadIdx.x + 256];
  float mean = block_sum256(v0 + v1, sm) * (1.f / EMB);
  float d0 = v0 - mean, d1 = v1 - mean;
  float var = block_sum256(d0 * d0 + d1 * d1, sm) * (1.f / EMB);
  float inv = rsqrtf(var + 1e-5f);
  bf16* yr = out + (size_t)t * EMB;
  yr[threadIdx.x]       = (bf16)(d0 * inv * g[threadIdx.x]       + b[threadIdx.x]);
  yr[threadIdx.x + 256] = (bf16)(d1 * inv * g[threadIdx.x + 256] + b[threadIdx.x + 256]);
}

// ------------------------------- WMMA GEMM ---------------------------------
// Y[m,n] = epilogue( sum_k A[m,k]*W[n,k] ), A:[M x K] bf16, W:[N x K] bf16.
// MODE 0: + bias (fp32 out)       MODE 1: + bias + res[m,n] (fp32 out)
// MODE 2: gelu(+bias) (bf16 out)  MODE 3: * scale_p[0]/sqrt(E) (fp32 out)
// block = 256 thr (8 waves); wave tile 16M x 64N; block tile 64M x 128N.
// K unrolled by 64: 20 b128 loads issued per iteration before 8 WMMAs.
template <int MODE, typename OutT>
__global__ __launch_bounds__(256)
void gemm_wmma_bf16(const bf16* __restrict__ A, const bf16* __restrict__ W,
                    const float* __restrict__ bias, const float* __restrict__ res,
                    const float* __restrict__ scale_p, OutT* __restrict__ Y,
                    int N, int K) {
  const int lane = threadIdx.x & 31;
  const int wave = threadIdx.x >> 5;
  const int mBase = blockIdx.y * 64 + (wave & 3) * 16;
  const int nBase = blockIdx.x * 128 + (wave >> 2) * 64;
  const int r  = lane & 15;
  const int hi = lane >> 4;

  v8f acc[4] = {};

  // per-lane base pointers (hoisted out of K loop)
  const bf16* Arow = A + (size_t)(mBase + r) * K + hi * 16;     // 16 contiguous K
  const bf16* Brow[4];
#pragma unroll
  for (int nt = 0; nt < 4; ++nt)
    Brow[nt] = W + (size_t)(nBase + nt * 16 + r) * K + hi * 16; // 16 contiguous K

  for (int k0 = 0; k0 < K; k0 += 64) {
    // issue all 20 b128 loads for two K-steps straight into operand regs
    v16bf a0 = load_frag16(Arow + k0);
    v16bf a1 = load_frag16(Arow + k0 + 32);
    v16bf b0[4], b1[4];
#pragma unroll
    for (int nt = 0; nt < 4; ++nt) {
      b0[nt] = load_frag16(Brow[nt] + k0);
      b1[nt] = load_frag16(Brow[nt] + k0 + 32);
    }
#pragma unroll
    for (int nt = 0; nt < 4; ++nt)
      acc[nt] = __builtin_amdgcn_wmma_f32_16x16x32_bf16(
          false, a0, false, b0[nt], (short)0, acc[nt], false, false);
#pragma unroll
    for (int nt = 0; nt < 4; ++nt)
      acc[nt] = __builtin_amdgcn_wmma_f32_16x16x32_bf16(
          false, a1, false, b1[nt], (short)0, acc[nt], false, false);
  }

  const float sc = (MODE == 3) ? scale_p[0] * 0.04419417382415922f : 0.f;
#pragma unroll
  for (int nt = 0; nt < 4; ++nt) {
    const int n = nBase + nt * 16 + r;
#pragma unroll
    for (int j = 0; j < 8; ++j) {
      const int m = mBase + j + hi * 8;       // C layout: VGPR j, lane half -> M
      float v = acc[nt][j];
      if (MODE != 3) v += bias[n];
      if (MODE == 2) v = gelu_exact(v);
      if (MODE == 1) v += res[(size_t)m * N + n];
      if (MODE == 3) v *= sc;
      Y[(size_t)m * N + n] = (OutT)v;
    }
  }
}

// ------------------------------- gates -------------------------------------
// block = 512 thr (16 waves) per token; wave = head. Two 512-dots per wave.
__global__ __launch_bounds__(512)
void gates_kernel(const bf16* __restrict__ xn,
                  const float* __restrict__ Wf, const float* __restrict__ bf,
                  const float* __restrict__ Ww, const float* __restrict__ bw,
                  const float* __restrict__ rarity,
                  float* __restrict__ gf, float* __restrict__ gw) {
  const int tok = blockIdx.x;
  const int h = threadIdx.x >> 5, lane = threadIdx.x & 31;
  const bf16* xr = xn + (size_t)tok * EMB;
  float sf = 0.f, sw = 0.f;
#pragma unroll
  for (int i = lane; i < EMB; i += 32) {
    float xv = (float)xr[i];
    sf += xv * Wf[h * EMB + i];
    sw += xv * Ww[h * EMB + i];
  }
  sf = wave_sum(sf);
  sw = wave_sum(sw);
  if (lane == 0) {
    float s1 = sigmoidf(sf + bf[h]);
    float gfv = 1.f - (1.f - s1) * (1.f - s1);
    float s2 = sigmoidf(sw + bw[h]);
    float gwv = s2 * s2;
    float rar = rarity[tok];
    gwv *= 0.5f + 0.5f * rar;
    gfv *= 0.9f + 0.1f * (1.f - rar);
    gf[tok * HEADS + h] = gfv;
    gw[tok * HEADS + h] = gwv;
  }
}

// --------------------------- k/q/v activations -----------------------------
// wave per (token, head): l2norm(k), l2norm(q), tanh(v) in place (fp32).
__global__ __launch_bounds__(256)
void act_kernel(float* __restrict__ k, float* __restrict__ q, float* __restrict__ v) {
  const int gwid = blockIdx.x * 8 + (threadIdx.x >> 5);   // 0 .. NTOK*HEADS-1
  const int lane = threadIdx.x & 31;
  const size_t base = (size_t)gwid * HDIM + lane;         // [tok][h*32+d] layout
  float kv = k[base];
  float kn = wave_sum(kv * kv);
  k[base] = kv / fmaxf(sqrtf(kn), 1e-12f);
  float qv = q[base];
  float qn = wave_sum(qv * qv);
  q[base] = qv / fmaxf(sqrtf(qn), 1e-12f);
  v[base] = tanhf(v[base]);
}

// ------------------------------ holo scan ----------------------------------
// block per (b,h); 1024 thr = 32 waves; wave = state row d; lane = column e.
// S_t = gf_t * S_{t-1} + gw_t * v_d * k_e ; readout_d = sum_e S * q_e.
// readout written as bf16 (feeds proj_out GEMM A-operand).
__global__ __launch_bounds__(1024)
void scan_kernel(const float* __restrict__ k, const float* __restrict__ q,
                 const float* __restrict__ v, const float* __restrict__ gf,
                 const float* __restrict__ gw, bf16* __restrict__ ro) {
  const int bh = blockIdx.x;
  const int b = bh >> 4, h = bh & 15;
  const int d = threadIdx.x >> 5;
  const int e = threadIdx.x & 31;
  float S = 0.f;
  for (int t = 0; t < SEQ; ++t) {
    const int tok = b * SEQ + t;
    const size_t hb = (size_t)tok * EMB + h * HDIM;
    float kf = k[hb + e];
    float qf = q[hb + e];
    float vf = v[hb + d];
    float f = gf[tok * HEADS + h];
    float w = gw[tok * HEADS + h];
    S = f * S + w * vf * kf;
    float r = wave_sum(S * qf);
    if (e == 0) ro[hb + d] = (bf16)r;
  }
}

// ------------------------------- launcher ----------------------------------

extern "C" void kernel_launch(void* const* d_in, const int* in_sizes, int n_in,
                              void* d_out, int out_size, void* d_ws, size_t ws_size,
                              hipStream_t stream) {
  (void)in_sizes; (void)n_in; (void)out_size; (void)ws_size;

  // -- inputs (setup_inputs dict insertion order, recursively flattened) --
  const int*   idx         = (const int*)d_in[0];
  const float* rarity      = (const float*)d_in[1];
  const float* center      = (const float*)d_in[2];
  const float* offset      = (const float*)d_in[3];
  const float* logit_scale = (const float*)d_in[4];
  const float* lnf_g       = (const float*)d_in[5];
  const float* lnf_b       = (const float*)d_in[6];
  const int L0 = 7, LS = 20;
  auto LP = [&](int l, int off) { return (const float*)d_in[L0 + l * LS + off]; };
  // per layer: 0 Wk 1 bk | 2 Wq 3 bq | 4 Wv 5 bv | 6 Wo 7 bo |
  //            8 Wgf 9 bgf | 10 Wgw 11 bgw | 12 ln1g 13 ln1b | 14 ln2g 15 ln2b |
  //            16 W1 17 b1 | 18 W2 19 b2

  // -- workspace carve-out --
  char* wsb = (char*)d_ws;
  auto carve = [&](size_t bytes) { void* p = wsb; wsb += (bytes + 255) & ~(size_t)255; return p; };
  float* x   = (float*)carve((size_t)NTOK * EMB * 4);
  float* kb  = (float*)carve((size_t)NTOK * EMB * 4);
  float* qb  = (float*)carve((size_t)NTOK * EMB * 4);
  float* vb  = (float*)carve((size_t)NTOK * EMB * 4);
  float* gf  = (float*)carve((size_t)NTOK * HEADS * 4);
  float* gw  = (float*)carve((size_t)NTOK * HEADS * 4);
  bf16*  xn  = (bf16*)carve((size_t)NTOK * EMB * 2);    // LN outputs (bf16)
  bf16*  ro  = (bf16*)carve((size_t)NTOK * EMB * 2);    // scan readout (bf16)
  bf16*  f1  = (bf16*)carve((size_t)NTOK * FFN * 2);    // FFN hidden (bf16)
  bf16*  cenb = (bf16*)carve((size_t)VOCAB * EMB * 2);  // center (bf16)
  bf16*  Wb[3][6];                                      // Wk Wq Wv Wo W1 W2 per layer
  for (int l = 0; l < 3; ++l) {
    Wb[l][0] = (bf16*)carve((size_t)EMB * EMB * 2);
    Wb[l][1] = (bf16*)carve((size_t)EMB * EMB * 2);
    Wb[l][2] = (bf16*)carve((size_t)EMB * EMB * 2);
    Wb[l][3] = (bf16*)carve((size_t)EMB * EMB * 2);
    Wb[l][4] = (bf16*)carve((size_t)FFN * EMB * 2);
    Wb[l][5] = (bf16*)carve((size_t)EMB * FFN * 2);
  }

  float* out = (float*)d_out;
  const dim3 blk256(256), blk512(512), blk1024(1024);

  // -- weight conversions (once per launch; deterministic) --
  auto cvt = [&](const float* src, bf16* dst, size_t n) {
    int n4 = (int)(n / 4);
    cvt_bf16_kernel<<<(n4 + 255) / 256, blk256, 0, stream>>>(src, dst, n4);
  };
  cvt(center, cenb, (size_t)VOCAB * EMB);
  for (int l = 0; l < 3; ++l) {
    cvt(LP(l, 0),  Wb[l][0], (size_t)EMB * EMB);
    cvt(LP(l, 2),  Wb[l][1], (size_t)EMB * EMB);
    cvt(LP(l, 4),  Wb[l][2], (size_t)EMB * EMB);
    cvt(LP(l, 6),  Wb[l][3], (size_t)EMB * EMB);
    cvt(LP(l, 16), Wb[l][4], (size_t)FFN * EMB);
    cvt(LP(l, 18), Wb[l][5], (size_t)EMB * FFN);
  }

  // embedding
  embed_kernel<<<(NTOK * EMB) / 256, blk256, 0, stream>>>(idx, center, offset, x);

  for (int l = 0; l < 3; ++l) {
    // LN1 -> xn (bf16)
    ln_kernel<<<NTOK, blk256, 0, stream>>>(x, xn, LP(l, 12), LP(l, 13));
    // QKV projections (N=512, K=512), fp32 out
    dim3 gP(EMB / 128, NTOK / 64);
    gemm_wmma_bf16<0, float><<<gP, blk256, 0, stream>>>(xn, Wb[l][0], LP(l, 1), nullptr, nullptr, kb, EMB, EMB);
    gemm_wmma_bf16<0, float><<<gP, blk256, 0, stream>>>(xn, Wb[l][1], LP(l, 3), nullptr, nullptr, qb, EMB, EMB);
    gemm_wmma_bf16<0, float><<<gP, blk256, 0, stream>>>(xn, Wb[l][2], LP(l, 5), nullptr, nullptr, vb, EMB, EMB);
    // gates
    gates_kernel<<<NTOK, blk512, 0, stream>>>(xn, LP(l, 8), LP(l, 9), LP(l, 10), LP(l, 11),
                                              rarity, gf, gw);
    // activations (l2norm k/q, tanh v)
    act_kernel<<<(NTOK * HEADS) / 8, blk256, 0, stream>>>(kb, qb, vb);
    // holographic scan -> ro (bf16)
    scan_kernel<<<BATCH * HEADS, blk1024, 0, stream>>>(kb, qb, vb, gf, gw, ro);
    // out projection + residual -> x (fp32)
    gemm_wmma_bf16<1, float><<<gP, blk256, 0, stream>>>(ro, Wb[l][3], LP(l, 7), x, nullptr, x, EMB, EMB);
    // LN2 -> xn (bf16)
    ln_kernel<<<NTOK, blk256, 0, stream>>>(x, xn, LP(l, 14), LP(l, 15));
    // FFN up + gelu (N=2048, K=512) -> f1 (bf16)
    dim3 gU(FFN / 128, NTOK / 64);
    gemm_wmma_bf16<2, bf16><<<gU, blk256, 0, stream>>>(xn, Wb[l][4], LP(l, 17), nullptr, nullptr, f1, FFN, EMB);
    // FFN down + residual (N=512, K=2048) -> x (fp32)
    gemm_wmma_bf16<1, float><<<gP, blk256, 0, stream>>>(f1, Wb[l][5], LP(l, 19), x, nullptr, x, EMB, FFN);
  }

  // final LN -> xn (bf16)
  ln_kernel<<<NTOK, blk256, 0, stream>>>(x, xn, lnf_g, lnf_b);
  // logits: [2048 x 512] @ center^T [512 x 32000] * (logit_scale/sqrt(512))
  dim3 gL(VOCAB / 128, NTOK / 64);
  gemm_wmma_bf16<3, float><<<gL, blk256, 0, stream>>>(xn, cenb, nullptr, nullptr, logit_scale,
                                                      out, VOCAB, EMB);
}
